// MultiHeadAttention_69561290326446
// MI455X (gfx1250) — compile-verified, matrix operand order checked by reading, not run on hardware
//
#include <hip/hip_runtime.h>

// ---------------------------------------------------------------------------
// MHA forward for MI455X (gfx1250, wave32, WMMA f32_16x16x32_f16).
// B=2, L=4096, DIM=512, H=8, dh=64. Outputs: out (B,L,DIM) then attention
// (B,H,L,L), concatenated f32 in d_out.
//
// Pipeline:
//   1) gemm_wt<float,M16>   : Q = (query@Wq^T + bq) * 1/sqrt(dh)  -> f16 ws
//   2) gemm_wt<float,M16>   : K = key@Wk^T + bk                   -> f16 ws
//   3) gemm_wt<float,MVT>   : Vt = (value@Wv^T + bv)^T            -> f16 ws (b, dim, L)
//   4) attn_fused           : per (b,h,16 q-rows): S=QK^T via WMMA into
//                             256KB LDS, softmax in-place, write attention,
//                             O = P@V via WMMA (split-K over 8 waves)
//   5) gemm_wt<_Float16,MF32>: out = O@Wo^T + bo                  -> f32 d_out
// ---------------------------------------------------------------------------

typedef __attribute__((ext_vector_type(16))) _Float16 v16h;
typedef __attribute__((ext_vector_type(8)))  _Float16 v8h;
typedef __attribute__((ext_vector_type(4)))  _Float16 v4h;
typedef __attribute__((ext_vector_type(8)))  float    v8f;
typedef __attribute__((ext_vector_type(4)))  float    v4f;

struct Frag {
  union { v16h v; v8h o[2]; v4h q[4]; _Float16 h[16]; };
};

static constexpr int BATCH = 2;
static constexpr int LSEQ  = 4096;
static constexpr int DIMC  = 512;
static constexpr int NH    = 8;
static constexpr int DH    = 64;
static constexpr int LDSS  = 40;   // LDS tile row stride in halves (80B: no bank conflicts, 16B aligned)

__device__ __forceinline__ v8f wmma16(const Frag& a, const Frag& b, v8f c) {
  return __builtin_amdgcn_wmma_f32_16x16x32_f16(
      /*neg_a=*/false, a.v, /*neg_b=*/false, b.v,
      /*c_mod=*/(short)0, c, /*reuse_a=*/false, /*reuse_b=*/false);
}

// A/B fragment (16x32 over K, f16). Lane l: row r=l&15, half=l>>4.
// Halves held (ISA 7.12.2): k in [hf*8, hf*8+8) then [16+hf*8, 16+hf*8+8).
__device__ __forceinline__ void frag_from_lds(Frag& f, const _Float16* base, int lane) {
  int r = lane & 15, hf = (lane >> 4) & 1;
  const _Float16* p = base + r * LDSS + hf * 8;
  f.o[0] = *(const v8h*)(p);
  f.o[1] = *(const v8h*)(p + 16);
}

// ---- global->LDS tile loaders (convert f32 -> f16 when needed) -------------
__device__ __forceinline__ void load16_to_lds(const float* src, _Float16* dst) {
#pragma unroll
  for (int i = 0; i < 4; ++i) {
    v4f f = *(const v4f*)(src + i * 4);
    v4h h;
#pragma unroll
    for (int j = 0; j < 4; ++j) h[j] = (_Float16)f[j];
    *(v4h*)(dst + i * 4) = h;
  }
}
__device__ __forceinline__ void load16_to_lds(const _Float16* src, _Float16* dst) {
  *(v8h*)(dst)     = *(const v8h*)(src);
  *(v8h*)(dst + 8) = *(const v8h*)(src + 8);
}
__device__ __forceinline__ void load8f_to_lds(const float* src, _Float16* dst) {
  v4f f0 = *(const v4f*)(src);
  v4f f1 = *(const v4f*)(src + 4);
  v4h h0, h1;
#pragma unroll
  for (int j = 0; j < 4; ++j) { h0[j] = (_Float16)f0[j]; h1[j] = (_Float16)f1[j]; }
  *(v4h*)(dst)     = h0;
  *(v4h*)(dst + 4) = h1;
}

// ---------------------------------------------------------------------------
// GEMM: C[M,512] = A[M,512] @ W[512,512]^T + bias, tile 128x64, 8 waves (2x2
// 16x16 WMMA tiles each), K-step 32.  MODE: 0 = f16 row-major (scale applied),
// 2 = f16 transposed per batch (Vt: (b, dim, L)), 3 = f32 row-major.
// ---------------------------------------------------------------------------
template <typename AT, int MODE>
__global__ __launch_bounds__(256) void gemm_wt_kernel(
    const AT* __restrict__ A, const float* __restrict__ W,
    const float* __restrict__ bias, void* __restrict__ Out, float scale) {
  __shared__ _Float16 As[128 * LDSS];
  __shared__ _Float16 Bs[64 * LDSS];
  const int K = DIMC;
  int tid = threadIdx.x, w = tid >> 5, lane = tid & 31;
  int m0 = blockIdx.x * 128, n0 = blockIdx.y * 64;
  int wm = (w >> 1) * 32, wn = (w & 1) * 32;

  v8f acc[2][2] = {};

  for (int k0 = 0; k0 < K; k0 += 32) {
    __syncthreads();
    {  // A tile: 128 rows x 32 k, each thread 16 elements
      int row = tid >> 1, seg = (tid & 1) * 16;
      load16_to_lds(A + (size_t)(m0 + row) * K + k0 + seg, As + row * LDSS + seg);
    }
    {  // W tile: 64 rows x 32 k, each thread 8 elements
      int row = tid >> 2, seg = (tid & 3) * 8;
      load8f_to_lds(W + (size_t)(n0 + row) * K + k0 + seg, Bs + row * LDSS + seg);
    }
    __syncthreads();
    Frag a0, a1, b0, b1;
    frag_from_lds(a0, As + (wm + 0)  * LDSS, lane);
    frag_from_lds(a1, As + (wm + 16) * LDSS, lane);
    frag_from_lds(b0, Bs + (wn + 0)  * LDSS, lane);
    frag_from_lds(b1, Bs + (wn + 16) * LDSS, lane);
    acc[0][0] = wmma16(a0, b0, acc[0][0]);
    acc[0][1] = wmma16(a0, b1, acc[0][1]);
    acc[1][0] = wmma16(a1, b0, acc[1][0]);
    acc[1][1] = wmma16(a1, b1, acc[1][1]);
  }

  // Epilogue. C/D layout: lane l -> col n=l&15; VGPR j -> row (l>>4)*8 + j.
  int n = lane & 15, hf = (lane >> 4) & 1;
#pragma unroll
  for (int ti = 0; ti < 2; ++ti) {
#pragma unroll
    for (int tj = 0; tj < 2; ++tj) {
      int col = n0 + wn + tj * 16 + n;
      float bv = bias[col];
#pragma unroll
      for (int j = 0; j < 8; ++j) {
        int m = m0 + wm + ti * 16 + hf * 8 + j;
        float val = (acc[ti][tj][j] + bv) * scale;
        if (MODE == 0) {
          ((_Float16*)Out)[(size_t)m * DIMC + col] = (_Float16)val;
        } else if (MODE == 2) {
          int bb = m >> 12, l = m & (LSEQ - 1);   // L = 4096
          ((_Float16*)Out)[((size_t)bb * DIMC + col) * LSEQ + l] = (_Float16)val;
        } else {
          ((float*)Out)[(size_t)m * DIMC + col] = val;
        }
      }
    }
  }
}

// ---------------------------------------------------------------------------
// Fused attention. grid = (L/16, B*H), block = 256 (8 waves).
// Dynamic LDS: sS[16][4096] f32 scores (256KB) + sPart[8][16][64] partials
// (32KB) + sInv[16] (64B)  ~= 288KB -> one workgroup per WGP (320KB LDS).
// ---------------------------------------------------------------------------
__global__ __launch_bounds__(256) void attn_fused_kernel(
    const _Float16* __restrict__ Qf, const _Float16* __restrict__ Kf,
    const _Float16* __restrict__ Vt, const unsigned char* __restrict__ mask,
    float* __restrict__ attn_out, _Float16* __restrict__ Of) {
  extern __shared__ char smem_raw[];
  float* sS    = (float*)smem_raw;                 // 16 * 4096
  float* sPart = sS + 16 * LSEQ;                   // 8 * 16 * 64
  float* sInv  = sPart + 8 * 16 * DH;              // 16

  int tid = threadIdx.x, w = tid >> 5, lane = tid & 31;
  int r16 = lane & 15, hf = (lane >> 4) & 1;
  int bh = blockIdx.y, b = bh >> 3, h = bh & 7;
  int q0 = blockIdx.x * 16;

  // ---- Q fragments (rows q0..q0+15, k = 0..63), scale already folded in ----
  Frag qa[2];
  {
    const _Float16* qp = Qf + (size_t)(b * LSEQ + q0 + r16) * DIMC + h * DH;
#pragma unroll
    for (int s = 0; s < 2; ++s) {
      const _Float16* p = qp + s * 32 + hf * 8;
      qa[s].o[0] = *(const v8h*)(p);
      qa[s].o[1] = *(const v8h*)(p + 16);
    }
  }

  // ---- S = Q @ K^T into LDS; each wave covers 32 of 256 key tiles ---------
  for (int kt = w; kt < LSEQ / 16; kt += 8) {
    int n0 = kt * 16;
    const _Float16* kp = Kf + (size_t)(b * LSEQ + n0 + r16) * DIMC + h * DH;
    Frag kb0, kb1;
    kb0.o[0] = *(const v8h*)(kp + hf * 8);
    kb0.o[1] = *(const v8h*)(kp + 16 + hf * 8);
    kb1.o[0] = *(const v8h*)(kp + 32 + hf * 8);
    kb1.o[1] = *(const v8h*)(kp + 48 + hf * 8);
    v8f acc = {};
    acc = wmma16(qa[0], kb0, acc);
    acc = wmma16(qa[1], kb1, acc);
    bool msk = mask[b * LSEQ + n0 + r16] != 0;
#pragma unroll
    for (int j = 0; j < 8; ++j)
      sS[(hf * 8 + j) * LSEQ + n0 + r16] = msk ? -1e18f : acc[j];
  }
  __syncthreads();

  // ---- softmax: wave w handles rows 2w, 2w+1. Keep unnormalized exp in LDS,
  //      fold 1/sum into attention write + final O scaling. -----------------
#pragma unroll
  for (int rr = 0; rr < 2; ++rr) {
    int r = w * 2 + rr;
    float* row = sS + r * LSEQ;
    float mx = -3.0e38f;
    for (int c = lane; c < LSEQ; c += 32) mx = fmaxf(mx, row[c]);
#pragma unroll
    for (int off = 16; off; off >>= 1) mx = fmaxf(mx, __shfl_xor(mx, off, 32));
    float sum = 0.f;
    for (int c = lane; c < LSEQ; c += 32) {
      float p = __expf(row[c] - mx);
      row[c] = p;
      sum += p;
    }
#pragma unroll
    for (int off = 16; off; off >>= 1) sum += __shfl_xor(sum, off, 32);
    float inv = 1.0f / sum;
    if (lane == 0) sInv[r] = inv;
    float* ap = attn_out + ((size_t)bh * LSEQ + q0 + r) * LSEQ;
    for (int c = lane; c < LSEQ; c += 32) ap[c] = row[c] * inv;  // coalesced
  }
  __syncthreads();

  // ---- O = P @ V, split-K: wave w owns k in [w*512, w*512+512) ------------
  v8f oacc[4] = {};
  int kbase = w * 512;
  for (int ks = 0; ks < 512; ks += 32) {
    int kb = kbase + ks;
    // A fragment from LDS probabilities (f32 -> f16 convert)
    Frag pa;
    const float* pr = sS + r16 * LSEQ + kb;
    v4f f0 = *(const v4f*)(pr + hf * 8);
    v4f f1 = *(const v4f*)(pr + hf * 8 + 4);
    v4f f2 = *(const v4f*)(pr + 16 + hf * 8);
    v4f f3 = *(const v4f*)(pr + 16 + hf * 8 + 4);
#pragma unroll
    for (int i = 0; i < 4; ++i) {
      pa.h[i]      = (_Float16)f0[i];
      pa.h[4 + i]  = (_Float16)f1[i];
      pa.h[8 + i]  = (_Float16)f2[i];
      pa.h[12 + i] = (_Float16)f3[i];
    }
#pragma unroll
    for (int nt = 0; nt < 4; ++nt) {
      const _Float16* vp =
          Vt + ((size_t)b * DIMC + h * DH + nt * 16 + r16) * LSEQ + kb;
      Frag vb;
      vb.o[0] = *(const v8h*)(vp + hf * 8);
      vb.o[1] = *(const v8h*)(vp + 16 + hf * 8);
      oacc[nt] = wmma16(pa, vb, oacc[nt]);
    }
  }
  // partials -> LDS
#pragma unroll
  for (int nt = 0; nt < 4; ++nt)
#pragma unroll
    for (int j = 0; j < 8; ++j)
      sPart[(w * 16 + hf * 8 + j) * DH + nt * 16 + r16] = oacc[nt][j];
  __syncthreads();

  // reduce 8 partials, normalize, store O (f16, merged-head layout)
  {
    int r = tid >> 4;            // 0..15
    int c0 = (tid & 15) * 4;     // 0..60
#pragma unroll
    for (int cc = 0; cc < 4; ++cc) {
      int c = c0 + cc;
      float s = 0.f;
#pragma unroll
      for (int ww = 0; ww < 8; ++ww) s += sPart[(ww * 16 + r) * DH + c];
      s *= sInv[r];
      Of[(size_t)(b * LSEQ + q0 + r) * DIMC + h * DH + c] = (_Float16)s;
    }
  }
}

// ---------------------------------------------------------------------------
extern "C" void kernel_launch(void* const* d_in, const int* in_sizes, int n_in,
                              void* d_out, int out_size, void* d_ws, size_t ws_size,
                              hipStream_t stream) {
  const float* query = (const float*)d_in[0];
  const float* key   = (const float*)d_in[1];
  const float* value = (const float*)d_in[2];
  const unsigned char* mask = (const unsigned char*)d_in[3];
  const float* Wq = (const float*)d_in[4];
  const float* bq = (const float*)d_in[5];
  const float* Wk = (const float*)d_in[6];
  const float* bk = (const float*)d_in[7];
  const float* Wv = (const float*)d_in[8];
  const float* bv = (const float*)d_in[9];
  const float* Wo = (const float*)d_in[10];
  const float* bo = (const float*)d_in[11];

  const size_t NTOK = (size_t)BATCH * LSEQ;         // 8192
  const size_t PROJ = NTOK * DIMC;                  // 4,194,304 halves (8 MB)

  _Float16* Qf = (_Float16*)d_ws;
  _Float16* Kf = Qf + PROJ;
  _Float16* Vt = Kf + PROJ;
  _Float16* Of = Vt + PROJ;

  float* out_p  = (float*)d_out;
  float* attn_p = out_p + PROJ;                     // attention after out

  dim3 gGemm(64, 8), blk(256);
  const float inv_sqrt_dh = 0.125f;                 // 1/sqrt(64)

  gemm_wt_kernel<float, 0><<<gGemm, blk, 0, stream>>>(query, Wq, bq, Qf, inv_sqrt_dh);
  gemm_wt_kernel<float, 0><<<gGemm, blk, 0, stream>>>(key,   Wk, bk, Kf, 1.0f);
  gemm_wt_kernel<float, 2><<<gGemm, blk, 0, stream>>>(value, Wv, bv, Vt, 1.0f);

  size_t smem = (size_t)16 * LSEQ * 4 + (size_t)8 * 16 * DH * 4 + 16 * 4;
  (void)hipFuncSetAttribute((const void*)attn_fused_kernel,
                            hipFuncAttributeMaxDynamicSharedMemorySize, (int)smem);
  attn_fused_kernel<<<dim3(LSEQ / 16, BATCH * NH), blk, smem, stream>>>(
      Qf, Kf, Vt, mask, attn_p, Of);

  gemm_wt_kernel<_Float16, 3><<<gGemm, blk, 0, stream>>>(Of, Wo, bo, out_p, 1.0f);
}
